// HistoryModelBlind_19224273617149
// MI455X (gfx1250) — compile-verified, matrix operand order checked by reading, not run on hardware
//
#include <hip/hip_runtime.h>
#include <math.h>

// ---------------------------------------------------------------------------
// MI455X (gfx1250): wave32, WMMA bf16 16x16x32 f32-accumulate, TDM staging.
// ---------------------------------------------------------------------------

typedef __bf16 v16bf __attribute__((ext_vector_type(16)));
typedef __bf16 v8bf  __attribute__((ext_vector_type(8)));
typedef float  v8f   __attribute__((ext_vector_type(8)));
typedef unsigned int u32x4 __attribute__((ext_vector_type(4)));
typedef int          i32x4 __attribute__((ext_vector_type(4)));
typedef int          i32x8 __attribute__((ext_vector_type(8)));

#define EDIM 512
#define HDIM 512
#define GDIM 2048   // 4*H
#define IMGD 2048
#define BSZ  64
#define TSEG 128
#define PP   6
#define THIST 192

#if defined(__has_builtin)
#if __has_builtin(__builtin_amdgcn_tensor_load_to_lds)
#define USE_TDM 1
#endif
#endif
#ifndef USE_TDM
#define USE_TDM 0
#endif
// therock/amdgpu-toolchain ships the TDM header and uses the 6-arg builtin;
// ROCm 7.2 (clang-22) uses the 5-arg form.
#if __has_include(<hip/amd_detail/amd_gfx1250_TDM.h>)
#define TDM_6ARG 1
#else
#define TDM_6ARG 0
#endif

__device__ __forceinline__ float sigf(float x) { return 1.0f / (1.0f + __expf(-x)); }

// ---------------------------------------------------------------------------
// Embedding gather + f32 -> bf16 convert.  One thread per 8 elements.
// ---------------------------------------------------------------------------
__global__ void gather_embed_bf16(const float* __restrict__ emb,
                                  const int*   __restrict__ toks,
                                  __bf16*      __restrict__ out,
                                  long n_tok)
{
  long i = (long)blockIdx.x * blockDim.x + threadIdx.x;
  long total = n_tok * (EDIM / 8);
  if (i >= total) return;
  long tp = i >> 6;            // 64 8-wide chunks per token row
  int  ch = (int)(i & 63);
  int  tok = toks[tp];
  const float4* s = (const float4*)(emb + (long)tok * EDIM + ch * 8);
  float4 a = s[0], b = s[1];
  v8bf r;
  r[0] = (__bf16)a.x; r[1] = (__bf16)a.y; r[2] = (__bf16)a.z; r[3] = (__bf16)a.w;
  r[4] = (__bf16)b.x; r[5] = (__bf16)b.y; r[6] = (__bf16)b.z; r[7] = (__bf16)b.w;
  *(v8bf*)(out + tp * EDIM + ch * 8) = r;
}

// f32 -> bf16 bulk convert (n must be a multiple of 8; all our buffers are)
__global__ void f32_to_bf16(const float* __restrict__ in,
                            __bf16*      __restrict__ out, long n)
{
  long i = ((long)blockIdx.x * blockDim.x + threadIdx.x) * 8;
  if (i >= n) return;
  float4 a = *(const float4*)(in + i);
  float4 b = *(const float4*)(in + i + 4);
  v8bf r;
  r[0] = (__bf16)a.x; r[1] = (__bf16)a.y; r[2] = (__bf16)a.z; r[3] = (__bf16)a.w;
  r[4] = (__bf16)b.x; r[5] = (__bf16)b.y; r[6] = (__bf16)b.z; r[7] = (__bf16)b.w;
  *(v8bf*)(out + i) = r;
}

__global__ void vec_add_f32(const float* __restrict__ a,
                            const float* __restrict__ b,
                            float* __restrict__ o, int n)
{
  int i = blockIdx.x * blockDim.x + threadIdx.x;
  if (i < n) o[i] = a[i] + b[i];
}

// ---------------------------------------------------------------------------
// Fully fused masked LSTM scan.
//   Workgroup = 16 sequences x full hidden state. 512 threads = 16 waves.
//   Wave w owns gate columns [w*32, w*32+32) in each of the 4 gate blocks
//   (8 C-fragments per wave).  c and f32 h stay in registers for all T steps;
//   bf16 h round-trips LDS for the next step's WMMA A operand.  The x_t tile
//   (16 rows x 1KB, global row stride T*1KB) is staged into LDS by the
//   Tensor Data Mover (one 2D descriptor, TENSORcnt-tracked).
// ---------------------------------------------------------------------------
__global__ __launch_bounds__(512)
void lstm_scan_wmma(const __bf16* __restrict__ x,     // [Nseq*T, E]
                    const int*    __restrict__ lens,  // [Nseq]
                    const __bf16* __restrict__ Wih,   // [G, E] row-major
                    const __bf16* __restrict__ Whh,   // [G, H] row-major
                    const float*  __restrict__ bsum,  // [G]  (b_ih + b_hh)
                    float*        __restrict__ h_out, // [Nseq, H]
                    int T)
{
  __shared__ v16bf x_sh[16 * 32];   // 16 rows x 512 bf16 = 16 KB
  __shared__ v16bf h_sh[16 * 32];   // 16 rows x 512 bf16 = 16 KB

  const int  tid  = threadIdx.x;
  const int  wave = tid >> 5;
  const int  lane = tid & 31;
  const int  lo   = lane & 15;
  const int  hi   = lane >> 4;
  const int  colbase = wave * 32;
  const long seq0 = (long)blockIdx.x * 16;

  // zero initial hidden state
  v16bf zv;
  #pragma unroll
  for (int e = 0; e < 16; ++e) zv[e] = (__bf16)0.0f;
  for (int i = tid; i < 16 * 32; i += 512) h_sh[i] = zv;

  int lenreg[8];
  #pragma unroll
  for (int j = 0; j < 8; ++j) lenreg[j] = lens[seq0 + j + 8 * hi];

  float c_reg[2][8], h_reg[2][8];
  #pragma unroll
  for (int t2 = 0; t2 < 2; ++t2)
    #pragma unroll
    for (int j = 0; j < 8; ++j) { c_reg[t2][j] = 0.0f; h_reg[t2][j] = 0.0f; }

  float biasv[4][2];
  #pragma unroll
  for (int g = 0; g < 4; ++g)
    #pragma unroll
    for (int t2 = 0; t2 < 2; ++t2)
      biasv[g][t2] = bsum[g * HDIM + colbase + t2 * 16 + lo];

#if USE_TDM
  // ---- TDM D# group 1: geometry, invariant across time steps -------------
  // data_size=2B; tensor_dim0=512; tensor_dim1=16 rows; tile=[512,16];
  // tensor_dim0_stride = T*512 elements (row pitch of the [Nseq*T, E] array).
  i32x8 g1;
  g1[0] = 0x00010000;            // workgroup_mask=0 | data_size=1 (2 bytes)
  g1[1] = (int)(EDIM << 16);     // tensor_dim0[15:0] in bits [63:48]
  g1[2] = (int)(16u << 16);      // tensor_dim1[15:0] in bits [95:80]
  g1[3] = (int)(EDIM << 16);     // tile_dim0 in bits [127:112]
  g1[4] = 16;                    // tile_dim1 in bits [143:128]
  g1[5] = (int)(T * EDIM);       // tensor_dim0_stride[31:0]
  g1[6] = 0;
  g1[7] = 0;
  i32x4 z4 = {0, 0, 0, 0};
#if TDM_6ARG
  i32x8 z8 = {0, 0, 0, 0, 0, 0, 0, 0};
#endif
  const unsigned lds_x = (unsigned)(size_t)(&x_sh[0]);   // LDS byte offset
  const unsigned long long gbase =
      (unsigned long long)(size_t)(x + seq0 * (long)T * EDIM);
#endif

  for (int t = 0; t < T; ++t) {
    // ---- stage x_t tile into LDS -----------------------------------------
#if USE_TDM
    if (wave == 0) {
      unsigned long long ga = gbase + (unsigned long long)t * (EDIM * 2);
      u32x4 g0;
      g0[0] = 1u;                                   // count=1, user descriptor
      g0[1] = lds_x;                                // lds_addr
      g0[2] = (unsigned)(ga & 0xFFFFFFFFu);         // global_addr[31:0]
      g0[3] = (unsigned)((ga >> 32) & 0x1FFFFFFu)   // global_addr[56:32]
              | 0x80000000u;                        // type=2 ("image")
#if TDM_6ARG
      __builtin_amdgcn_tensor_load_to_lds(g0, g1, z4, z4, z8, 0);
#else
      __builtin_amdgcn_tensor_load_to_lds(g0, g1, z4, z4, 0);
#endif
      __builtin_amdgcn_s_wait_tensorcnt(0);
    }
#else
    {
      const v16bf* src =
          (const v16bf*)(x + ((seq0 + wave) * (long)T + t) * EDIM) + lane;
      x_sh[wave * 32 + lane] = *src;
    }
#endif
    __syncthreads();   // x visible; previous step's h writes visible

    v8f acc[4][2];
    #pragma unroll
    for (int g = 0; g < 4; ++g)
      #pragma unroll
      for (int t2 = 0; t2 < 2; ++t2) {
        v8f a;
        #pragma unroll
        for (int j = 0; j < 8; ++j) a[j] = biasv[g][t2];
        acc[g][t2] = a;
      }

    // input-gate contribution: K over E
    #pragma unroll 1
    for (int k = 0; k < EDIM; k += 32) {
      v16bf afr = x_sh[lo * 32 + (k >> 4) + hi];
      #pragma unroll
      for (int g = 0; g < 4; ++g)
        #pragma unroll
        for (int t2 = 0; t2 < 2; ++t2) {
          int col = g * HDIM + colbase + t2 * 16 + lo;
          v16bf bfr = *(const v16bf*)(Wih + (long)col * EDIM + k + (hi << 4));
          acc[g][t2] = __builtin_amdgcn_wmma_f32_16x16x32_bf16(
              false, afr, false, bfr, (short)0, acc[g][t2], false, false);
        }
    }
    // recurrent contribution: K over H
    #pragma unroll 1
    for (int k = 0; k < HDIM; k += 32) {
      v16bf afr = h_sh[lo * 32 + (k >> 4) + hi];
      #pragma unroll
      for (int g = 0; g < 4; ++g)
        #pragma unroll
        for (int t2 = 0; t2 < 2; ++t2) {
          int col = g * HDIM + colbase + t2 * 16 + lo;
          v16bf bfr = *(const v16bf*)(Whh + (long)col * HDIM + k + (hi << 4));
          acc[g][t2] = __builtin_amdgcn_wmma_f32_16x16x32_bf16(
              false, afr, false, bfr, (short)0, acc[g][t2], false, false);
        }
    }
    __syncthreads();   // all waves done reading h_sh / x_sh

    // gate nonlinearities + masked state update (i,f,g,o torch order)
    #pragma unroll
    for (int t2 = 0; t2 < 2; ++t2) {
      #pragma unroll
      for (int j = 0; j < 8; ++j) {
        if (t < lenreg[j]) {
          float gi = acc[0][t2][j], gf = acc[1][t2][j];
          float gg = acc[2][t2][j], go = acc[3][t2][j];
          float cn = sigf(gf) * c_reg[t2][j] + sigf(gi) * tanhf(gg);
          float hn = sigf(go) * tanhf(cn);
          c_reg[t2][j] = cn;
          h_reg[t2][j] = hn;
          int row = j + 8 * hi;
          int col = colbase + t2 * 16 + lo;
          ((__bf16*)h_sh)[row * HDIM + col] = (__bf16)hn;
        }
      }
    }
    __syncthreads();   // h writes complete before next step's reads
  }

  // write final f32 hidden state (each lane owns 16 (row,col) elements)
  #pragma unroll
  for (int t2 = 0; t2 < 2; ++t2)
    #pragma unroll
    for (int j = 0; j < 8; ++j) {
      int row = j + 8 * hi;
      int col = colbase + t2 * 16 + lo;
      h_out[(seq0 + row) * HDIM + col] = h_reg[t2][j];
    }
}

// ---------------------------------------------------------------------------
// Fused epilogue: sep = relu(simg @ w_sep^T + b_sep + h_hist);
//                 out = (sep / max(||sep||,1e-12)) . h_seg
// Workgroup = 16 rows (b,p pairs), 16 waves covering the 512 hidden columns.
// ---------------------------------------------------------------------------
__global__ __launch_bounds__(512)
void sep_norm_dot_wmma(const __bf16* __restrict__ simg,   // [384, IMG] bf16
                       const __bf16* __restrict__ Wsep,   // [H, IMG] bf16
                       const float*  __restrict__ bsep,   // [H]
                       const float*  __restrict__ h_hist, // [384, H]
                       const float*  __restrict__ h_seg,  // [64, H]
                       float*        __restrict__ out)    // [384]
{
  __shared__ float norm_sh[16];
  __shared__ float dot_sh[16];

  const int  tid  = threadIdx.x;
  const int  wave = tid >> 5;
  const int  lane = tid & 31;
  const int  lo   = lane & 15;
  const int  hi   = lane >> 4;
  const int  colbase = wave * 32;
  const long row0 = (long)blockIdx.x * 16;

  if (tid < 16) { norm_sh[tid] = 0.0f; dot_sh[tid] = 0.0f; }
  __syncthreads();

  v8f acc[2];
  #pragma unroll
  for (int t2 = 0; t2 < 2; ++t2) {
    float b = bsep[colbase + t2 * 16 + lo];
    #pragma unroll
    for (int j = 0; j < 8; ++j) acc[t2][j] = b;
  }

  #pragma unroll 1
  for (int k = 0; k < IMGD; k += 32) {
    // A fragment straight from global (L2/L0 resident; 16 waves share rows)
    v16bf afr = *(const v16bf*)(simg + (row0 + lo) * (long)IMGD + k + (hi << 4));
    #pragma unroll
    for (int t2 = 0; t2 < 2; ++t2) {
      int col = colbase + t2 * 16 + lo;
      v16bf bfr = *(const v16bf*)(Wsep + (long)col * IMGD + k + (hi << 4));
      acc[t2] = __builtin_amdgcn_wmma_f32_16x16x32_bf16(
          false, afr, false, bfr, (short)0, acc[t2], false, false);
    }
  }

  float v[2][8];
  #pragma unroll
  for (int t2 = 0; t2 < 2; ++t2)
    #pragma unroll
    for (int j = 0; j < 8; ++j) {
      int  row  = j + 8 * hi;
      long rowg = row0 + row;
      int  col  = colbase + t2 * 16 + lo;
      float s = acc[t2][j] + h_hist[rowg * HDIM + col];
      s = fmaxf(s, 0.0f);
      v[t2][j] = s;
      atomicAdd(&norm_sh[row], s * s);   // ds_add_f32
    }
  __syncthreads();

  #pragma unroll
  for (int t2 = 0; t2 < 2; ++t2)
    #pragma unroll
    for (int j = 0; j < 8; ++j) {
      int  row  = j + 8 * hi;
      long rowg = row0 + row;
      int  col  = colbase + t2 * 16 + lo;
      long b    = rowg / PP;
      atomicAdd(&dot_sh[row], v[t2][j] * h_seg[b * HDIM + col]);
    }
  __syncthreads();

  if (tid < 16)
    out[row0 + tid] = dot_sh[tid] / fmaxf(sqrtf(norm_sh[tid]), 1e-12f);
}

// ---------------------------------------------------------------------------
// Host-side orchestration
// ---------------------------------------------------------------------------
extern "C" void kernel_launch(void* const* d_in, const int* in_sizes, int n_in,
                              void* d_out, int out_size, void* d_ws, size_t ws_size,
                              hipStream_t stream)
{
  (void)in_sizes; (void)n_in; (void)out_size; (void)ws_size;

  const int*   segment      = (const int*)  d_in[0];   // [64,128]
  const int*   lengths      = (const int*)  d_in[1];   // [64]
  const int*   hist_tokens  = (const int*)  d_in[2];   // [64,6,192]
  const int*   hist_lengths = (const int*)  d_in[3];   // [64,6]
  const float* simg_f32     = (const float*)d_in[4];   // [64,6,2048]
  /* d_in[5] visual_context is unused by the reference */
  const float* emb          = (const float*)d_in[6];   // [30000,512]
  const float* w_ih         = (const float*)d_in[7];   // [2048,512]
  const float* w_hh         = (const float*)d_in[8];   // [2048,512]
  const float* b_ih         = (const float*)d_in[9];
  const float* b_hh         = (const float*)d_in[10];
  const float* wh_ih        = (const float*)d_in[11];
  const float* wh_hh        = (const float*)d_in[12];
  const float* bh_ih        = (const float*)d_in[13];
  const float* bh_hh        = (const float*)d_in[14];
  const float* w_sep        = (const float*)d_in[15];  // [512,2048]
  const float* b_sep        = (const float*)d_in[16];  // [512]
  float*       out          = (float*)d_out;           // [64*6]

  // workspace carve-out (256B aligned slabs)
  size_t off = 0;
  auto take = [&](size_t bytes) -> void* {
    void* p = (void*)((char*)d_ws + off);
    off += (bytes + 255) & ~(size_t)255;
    return p;
  };
  const long n_seg_tok  = (long)BSZ * TSEG;            // 8192
  const long n_hist_tok = (long)BSZ * PP * THIST;      // 73728

  __bf16* x_seg    = (__bf16*)take(n_seg_tok  * EDIM * 2);
  __bf16* x_hist   = (__bf16*)take(n_hist_tok * EDIM * 2);
  __bf16* wih_bf   = (__bf16*)take((size_t)GDIM * EDIM * 2);
  __bf16* whh_bf   = (__bf16*)take((size_t)GDIM * HDIM * 2);
  __bf16* whih_bf  = (__bf16*)take((size_t)GDIM * EDIM * 2);
  __bf16* whhh_bf  = (__bf16*)take((size_t)GDIM * HDIM * 2);
  __bf16* wsep_bf  = (__bf16*)take((size_t)HDIM * IMGD * 2);
  __bf16* simg_bf  = (__bf16*)take((size_t)BSZ * PP * IMGD * 2);
  float*  bsum_seg = (float*)take(GDIM * 4);
  float*  bsum_h   = (float*)take(GDIM * 4);
  float*  h_seg    = (float*)take((size_t)BSZ * HDIM * 4);
  float*  h_hist   = (float*)take((size_t)BSZ * PP * HDIM * 4);

  // 1) embedding gathers (f32 -> bf16)
  {
    long tot = n_seg_tok * (EDIM / 8);
    gather_embed_bf16<<<(int)((tot + 255) / 256), 256, 0, stream>>>(
        emb, segment, x_seg, n_seg_tok);
    tot = n_hist_tok * (EDIM / 8);
    gather_embed_bf16<<<(int)((tot + 255) / 256), 256, 0, stream>>>(
        emb, hist_tokens, x_hist, n_hist_tok);
  }
  // 2) weight / image conversions
  {
    long nw = (long)GDIM * EDIM;                       // 1048576
    int  bl = (int)((nw / 8 + 255) / 256);
    f32_to_bf16<<<bl, 256, 0, stream>>>(w_ih,  wih_bf,  nw);
    f32_to_bf16<<<bl, 256, 0, stream>>>(w_hh,  whh_bf,  nw);
    f32_to_bf16<<<bl, 256, 0, stream>>>(wh_ih, whih_bf, nw);
    f32_to_bf16<<<bl, 256, 0, stream>>>(wh_hh, whhh_bf, nw);
    f32_to_bf16<<<bl, 256, 0, stream>>>(w_sep, wsep_bf, (long)HDIM * IMGD);
    long ns = (long)BSZ * PP * IMGD;                   // 786432
    f32_to_bf16<<<(int)((ns / 8 + 255) / 256), 256, 0, stream>>>(
        simg_f32, simg_bf, ns);
    vec_add_f32<<<(GDIM + 255) / 256, 256, 0, stream>>>(b_ih,  b_hh,  bsum_seg, GDIM);
    vec_add_f32<<<(GDIM + 255) / 256, 256, 0, stream>>>(bh_ih, bh_hh, bsum_h,   GDIM);
  }
  // 3) fused LSTM scans (WMMA + TDM staging)
  lstm_scan_wmma<<<BSZ / 16, 512, 0, stream>>>(
      x_seg, lengths, wih_bf, whh_bf, bsum_seg, h_seg, TSEG);
  lstm_scan_wmma<<<(BSZ * PP) / 16, 512, 0, stream>>>(
      x_hist, hist_lengths, whih_bf, whhh_bf, bsum_h, h_hist, THIST);
  // 4) fused projection + relu + normalize + dot (WMMA + LDS reductions)
  sep_norm_dot_wmma<<<(BSZ * PP) / 16, 512, 0, stream>>>(
      simg_bf, wsep_bf, b_sep, h_hist, h_seg, out);
}
